// DeformationNetworkGraphConvolutionalFullRes_44873818308895
// MI455X (gfx1250) — matrix operand order repaired and checked by reading, not compiled, not run on hardware
//
#include <hip/hip_runtime.h>
#include <hip/hip_bf16.h>

// ---------------------------------------------------------------------------
// Problem constants (from the reference)
// ---------------------------------------------------------------------------
#define BB      4
#define VV      10242
#define NN      (BB * VV)          // 40968
#define NP      40976              // NN padded to multiple of 16
#define K1      3840               // sampled feature channels
#define HID     128
#define CONCAT  387                // 128 + 3 + 256
#define K2P     416                // CONCAT padded to multiple of 32
#define EDGES   (BB * 30720)       // 122880
#define TILE_ELEMS 4096            // one 32-K B tile: 8 nt * 2 half * 32 lane * 8 = 4096 bf16 (8KB)

typedef __attribute__((ext_vector_type(16))) __bf16 v16bf;
typedef __attribute__((ext_vector_type(8)))  __bf16 v8bf;
typedef __attribute__((ext_vector_type(8)))  float  v8f;

// ---------------------------------------------------------------------------
// Pack a row-major f32 weight matrix W[Ksrc x 128] into a WMMA B-fragment
// layout, half-major so LDS reads use 16B lane stride (2-way banks, optimal):
//   index = (((kt*8 + nt)*2 + h)*32 + lane)*8 + j
//   value = W[k][n], n = nt*16 + (lane&15),
//                    k = kt*32 + h*16 + (lane>=16 ? 8 : 0) + j
// K zero-padded from Ksrc to Kpad (multiple of 32).
// ---------------------------------------------------------------------------
__global__ void pack_b_kernel(const float* __restrict__ W, __bf16* __restrict__ P,
                              int Ksrc, int Kpad) {
    int t = blockIdx.x * blockDim.x + threadIdx.x;
    int total = Kpad * HID;
    if (t >= total) return;
    int j  = t & 7;
    int l  = (t >> 3) & 31;
    int h  = (t >> 8) & 1;
    int nt = (t >> 9) & 7;
    int kt = t >> 12;
    int n  = nt * 16 + (l & 15);
    int k  = kt * 32 + h * 16 + ((l >> 4) & 1) * 8 + j;
    float v = (k < Ksrc) ? W[(size_t)k * HID + n] : 0.0f;
    P[t] = (__bf16)v;
}

// ---------------------------------------------------------------------------
// Bilinear sampling of one feature map for one vertex row (align_corners=true)
// ---------------------------------------------------------------------------
__device__ __forceinline__ void sample_fm(const float* __restrict__ fm, int C, int W,
                                          int b, float gx, float gy,
                                          __bf16* __restrict__ out,
                                          int tid, int nthreads) {
    float x = (gx + 1.0f) * 0.5f * (float)(W - 1);
    float y = (gy + 1.0f) * 0.5f * (float)(W - 1);
    float x0f = floorf(x), y0f = floorf(y);
    float wx1 = x - x0f, wy1 = y - y0f;
    float wx0 = 1.0f - wx1, wy0 = 1.0f - wy1;
    int x0 = min(max((int)x0f, 0), W - 1);
    int x1 = min(max((int)x0f + 1, 0), W - 1);
    int y0 = min(max((int)y0f, 0), W - 1);
    int y1 = min(max((int)y0f + 1, 0), W - 1);
    float w00 = wy0 * wx0, w01 = wy0 * wx1, w10 = wy1 * wx0, w11 = wy1 * wx1;
    size_t plane = (size_t)W * W;
    size_t base  = (size_t)b * C * plane;
    int o00 = y0 * W + x0, o01 = y0 * W + x1, o10 = y1 * W + x0, o11 = y1 * W + x1;
    for (int c = tid; c < C; c += nthreads) {
        const float* p = fm + base + (size_t)c * plane;
        float v = w00 * p[o00] + w01 * p[o01] + w10 * p[o10] + w11 * p[o11];
        out[c] = (__bf16)v;
    }
}

// One block per output row; writes bf16 row of 3840 channels (zeros for pad rows).
__global__ void sampler_kernel(const float* __restrict__ f1, const float* __restrict__ f2,
                               const float* __restrict__ f3, const float* __restrict__ f4,
                               const float* __restrict__ averts,
                               __bf16* __restrict__ samp, int m_start) {
    int r = blockIdx.x;
    int m = m_start + r;
    __bf16* out = samp + (size_t)r * K1;
    int tid = threadIdx.x, nt = blockDim.x;
    if (m >= NN) {                       // pad rows -> zeros
        for (int k = tid; k < K1; k += nt) out[k] = (__bf16)0.0f;
        return;
    }
    int b = m / VV, v = m - b * VV;
    float gx = averts[((size_t)b * VV + v) * 3 + 0];
    float gy = averts[((size_t)b * VV + v) * 3 + 1];
    sample_fm(f1,  256, 56, b, gx, gy, out,        tid, nt);
    sample_fm(f2,  512, 28, b, gx, gy, out +  256, tid, nt);
    sample_fm(f3, 1024, 14, b, gx, gy, out +  768, tid, nt);
    sample_fm(f4, 2048,  7, b, gx, gy, out + 1792, tid, nt);
}

// ---------------------------------------------------------------------------
// WMMA GEMM with LDS-staged, double-buffered B tiles.
//   - 256 threads = 8 waves; wave w computes the 16-row strip blockIdx.x*8+w.
//   - Per 32-K step, the block cooperatively stages one 8KB B tile into LDS
//     (each thread: 2 x 16B), all 8 waves consume it via ds_load_b128.
//   - Double buffer: stores always target the buffer not being read this
//     iteration; the previous barrier fenced its prior readers.
//   - Out-of-range waves clamp to the last strip (EXEC all-ones for WMMA)
//     and skip only the final store, so they still help stage B.
// A-fragment (16-bit A 16x32): lane 0-15 rows M=0..15 hold K {0..7,16..23},
// lanes 16-31 hold K {8..15,24..31} -> two contiguous 16B loads per lane.
// ---------------------------------------------------------------------------
template <bool RELU, bool OUT_BF16>
__global__ void wmma_gemm_kernel(const __bf16* __restrict__ A, int lda,
                                 const __bf16* __restrict__ Bp,
                                 const float* __restrict__ bias,
                                 int K, int mTiles,
                                 float* __restrict__ outF,
                                 __bf16* __restrict__ outB, int ldo) {
    __shared__ __bf16 Bt[2][TILE_ELEMS];         // 2 x 8KB

    const int tid   = threadIdx.x;
    const int lane  = tid & 31;
    const int strip = blockIdx.x * 8 + (tid >> 5);
    const bool active = strip < mTiles;
    const int cstrip  = active ? strip : (mTiles - 1);
    const int hi   = lane >> 4;              // 0: K+0/+16 half, 1: K+8/+24 half
    const int mrow = cstrip * 16 + (lane & 15);

    const __bf16* arow = A + (size_t)mrow * lda + hi * 8;
    const uint4*  bsrc = (const uint4*)Bp;   // 16B chunks; 512 per tile

    v8f acc[8];
#pragma unroll
    for (int i = 0; i < 8; ++i) acc[i] = (v8f){0.f,0.f,0.f,0.f,0.f,0.f,0.f,0.f};

    const int ktiles = K >> 5;

    // Preload tile 0 into buffer 0.
    {
        uint4 r0 = bsrc[tid];
        uint4 r1 = bsrc[tid + 256];
        uint4* dst = (uint4*)&Bt[0][0];
        dst[tid]       = r0;
        dst[tid + 256] = r1;
    }
    __syncthreads();

    for (int kt = 0; kt < ktiles; ++kt) {
        const int cur = kt & 1;

        // Begin fetching next B tile into registers (overlaps with WMMA).
        uint4 r0, r1;
        const bool more = (kt + 1) < ktiles;
        if (more) {
            const uint4* src = bsrc + (size_t)(kt + 1) * (TILE_ELEMS / 8);
            r0 = src[tid];
            r1 = src[tid + 256];
            __builtin_prefetch(arow + (size_t)(kt + 1) * 32, 0, 1); // next A tile
        }

        // A fragment: two 16-byte global loads per lane.
        v8bf a0 = *(const v8bf*)(arow + (size_t)kt * 32);
        v8bf a1 = *(const v8bf*)(arow + (size_t)kt * 32 + 16);
        v16bf a;
#pragma unroll
        for (int i = 0; i < 8; ++i) { a[i] = a0[i]; a[8 + i] = a1[i]; }

        // 8 WMMAs fed from LDS (ds_load_b128, 16B lane stride).
        const __bf16* bbase = &Bt[cur][0];
#pragma unroll
        for (int nt = 0; nt < 8; ++nt) {
            v8bf blo = *(const v8bf*)(bbase + ((nt * 2 + 0) * 32 + lane) * 8);
            v8bf bhi = *(const v8bf*)(bbase + ((nt * 2 + 1) * 32 + lane) * 8);
            v16bf b;
#pragma unroll
            for (int i = 0; i < 8; ++i) { b[i] = blo[i]; b[8 + i] = bhi[i]; }
            acc[nt] = __builtin_amdgcn_wmma_f32_16x16x32_bf16(
                false, a, false, b, (short)0, acc[nt], false, false);
        }

        // Store next tile into the buffer NOT read this iteration.
        if (more) {
            uint4* dst = (uint4*)&Bt[cur ^ 1][0];
            dst[tid]       = r0;
            dst[tid + 256] = r1;
        }
        __syncthreads();
    }

    if (!active) return;

    // Epilogue. C/D layout: lane l, VGPR r -> M = r + (l>=16 ? 8 : 0), N = l&15.
    const int mbase = strip * 16 + hi * 8;
    const int ncol  = lane & 15;
#pragma unroll
    for (int nt = 0; nt < 8; ++nt) {
        int n = nt * 16 + ncol;
        float bv = bias[n];
#pragma unroll
        for (int r = 0; r < 8; ++r) {
            float v = acc[nt][r] + bv;
            if (RELU) v = fmaxf(v, 0.0f);
            size_t idx = (size_t)(mbase + r) * ldo + n;
            if (OUT_BF16) outB[idx] = (__bf16)v;
            else          outF[idx] = v;
        }
    }
}

// ---------------------------------------------------------------------------
// Fill x columns 128..415: [verts_packed(3) | image_enc(256) | zero-pad(29)]
// ---------------------------------------------------------------------------
__global__ void concat_fill_kernel(const float* __restrict__ verts,
                                   const float* __restrict__ enc,
                                   __bf16* __restrict__ xb) {
    int m = blockIdx.x;                  // [0, NP)
    int col = 128 + threadIdx.x;         // 288 threads -> cols 128..415
    float v = 0.0f;
    if (m < NN) {
        int b = m / VV;
        if (col < 131)      v = verts[(size_t)m * 3 + (col - 128)];
        else if (col < 387) v = enc[(size_t)b * 256 + (col - 131)];
    }
    xb[(size_t)m * K2P + col] = (__bf16)v;
}

// Undirected scatter-add: y[e0] += h1[e1]; y[e1] += h1[e0]  (per channel)
__global__ void edge_scatter_kernel(const int* __restrict__ edges,
                                    const float* __restrict__ h1,
                                    float* __restrict__ y) {
    int e = blockIdx.x;
    int c = threadIdx.x;                 // 128
    int a = edges[2 * e + 0];
    int b = edges[2 * e + 1];
    float va = h1[(size_t)a * HID + c];
    float vb = h1[(size_t)b * HID + c];
    atomicAdd(&y[(size_t)a * HID + c], vb);
    atomicAdd(&y[(size_t)b * HID + c], va);
}

// ReLU, then write bf16 into x (stride K2P) and f32 mirror (for final proj)
__global__ void relu_store_kernel(const float* __restrict__ y,
                                  __bf16* __restrict__ xb,
                                  float* __restrict__ xf) {
    size_t t = (size_t)blockIdx.x * blockDim.x + threadIdx.x;
    if (t >= (size_t)NP * HID) return;
    int m = (int)(t >> 7);
    int c = (int)(t & 127);
    float v = fmaxf(y[t], 0.0f);
    xb[(size_t)m * K2P + c] = (__bf16)v;
    xf[t] = v;
}

// out[m, 0..2] = xf[m, :] @ off_w + off_b
__global__ void final_proj_kernel(const float* __restrict__ xf,
                                  const float* __restrict__ offw,
                                  const float* __restrict__ offb,
                                  float* __restrict__ out) {
    int m = blockIdx.x * blockDim.x + threadIdx.x;
    if (m >= NN) return;
    float s0 = offb[0], s1 = offb[1], s2 = offb[2];
    const float* row = xf + (size_t)m * HID;
    for (int c = 0; c < HID; ++c) {
        float v = row[c];
        s0 = fmaf(v, offw[c * 3 + 0], s0);
        s1 = fmaf(v, offw[c * 3 + 1], s1);
        s2 = fmaf(v, offw[c * 3 + 2], s2);
    }
    out[(size_t)m * 3 + 0] = s0;
    out[(size_t)m * 3 + 1] = s1;
    out[(size_t)m * 3 + 2] = s2;
}

// ---------------------------------------------------------------------------
// Host-side orchestration
// ---------------------------------------------------------------------------
extern "C" void kernel_launch(void* const* d_in, const int* in_sizes, int n_in,
                              void* d_out, int out_size, void* d_ws, size_t ws_size,
                              hipStream_t stream) {
    const float* f1    = (const float*)d_in[0];
    const float* f2    = (const float*)d_in[1];
    const float* f3    = (const float*)d_in[2];
    const float* f4    = (const float*)d_in[3];
    const float* av    = (const float*)d_in[4];   // aligned_verts [B,V,3]
    const float* vp    = (const float*)d_in[5];   // verts_packed [N,3]
    const float* enc   = (const float*)d_in[6];   // image_enc [B,256]
    const int*   edges = (const int*)d_in[7];     // [E,2]
    const float* bw    = (const float*)d_in[8];   // bottleneck_w [3840,128]
    const float* bb    = (const float*)d_in[9];
    const float* g0w0  = (const float*)d_in[10];
    const float* g0b0  = (const float*)d_in[11];
    const float* g0w1  = (const float*)d_in[12];
    const float* g0b1  = (const float*)d_in[13];
    const float* gw0   = (const float*)d_in[14];  // [7,128,128]
    const float* gb0   = (const float*)d_in[15];
    const float* gw1   = (const float*)d_in[16];
    const float* gb1   = (const float*)d_in[17];
    const float* offw  = (const float*)d_in[18];
    const float* offb  = (const float*)d_in[19];
    float* out = (float*)d_out;
    (void)n_in; (void)out_size; (void)in_sizes;

    // --- workspace layout (256B aligned slots) ---
    char* ws = (char*)d_ws;
    size_t off = 0;
    auto take = [&](size_t bytes) -> char* {
        char* p = ws + off;
        off = (off + bytes + 255) & ~(size_t)255;
        return p;
    };
    __bf16* xb    = (__bf16*)take((size_t)NP * K2P * sizeof(__bf16));  // x, bf16, stride 416
    float*  xf    = (float*) take((size_t)NP * HID * sizeof(float));   // x, f32 mirror
    float*  Y     = (float*) take((size_t)NP * HID * sizeof(float));   // h0 / accumulator
    float*  H1    = (float*) take((size_t)NP * HID * sizeof(float));   // h1
    __bf16* Pbot  = (__bf16*)take((size_t)K1  * HID * sizeof(__bf16));
    __bf16* Pg0w0 = (__bf16*)take((size_t)K2P * HID * sizeof(__bf16));
    __bf16* Pg0w1 = (__bf16*)take((size_t)K2P * HID * sizeof(__bf16));
    __bf16* Pgw0  = (__bf16*)take((size_t)7 * HID * HID * sizeof(__bf16));
    __bf16* Pgw1  = (__bf16*)take((size_t)7 * HID * HID * sizeof(__bf16));
    size_t fixed = off;
    __bf16* samp  = (__bf16*)(ws + fixed);

    // Chunk the sampled-feature buffer through remaining workspace
    // (deterministic for a given ws_size -> graph-capture safe).
    size_t avail = (ws_size > fixed) ? (ws_size - fixed) : 0;
    long long chunkRows = (long long)(avail / ((size_t)K1 * sizeof(__bf16)));
    chunkRows &= ~15LL;
    if (chunkRows < 16) chunkRows = 16;
    if (chunkRows > NP) chunkRows = NP;

    // --- pack weights into WMMA B-fragment layout ---
    {
        int tot = K1 * HID;
        pack_b_kernel<<<(tot + 255) / 256, 256, 0, stream>>>(bw, Pbot, K1, K1);
        tot = K2P * HID;
        pack_b_kernel<<<(tot + 255) / 256, 256, 0, stream>>>(g0w0, Pg0w0, CONCAT, K2P);
        pack_b_kernel<<<(tot + 255) / 256, 256, 0, stream>>>(g0w1, Pg0w1, CONCAT, K2P);
        tot = HID * HID;
        for (int i = 0; i < 7; ++i) {
            pack_b_kernel<<<(tot + 255) / 256, 256, 0, stream>>>(
                gw0 + (size_t)i * HID * HID, Pgw0 + (size_t)i * HID * HID, HID, HID);
            pack_b_kernel<<<(tot + 255) / 256, 256, 0, stream>>>(
                gw1 + (size_t)i * HID * HID, Pgw1 + (size_t)i * HID * HID, HID, HID);
        }
    }

    // --- x columns 128..415: verts | image_enc | zero pad ---
    concat_fill_kernel<<<NP, 288, 0, stream>>>(vp, enc, xb);

    // --- vert_align + bottleneck GEMM (chunked), writes x cols 0..127 ---
    for (long long ms = 0; ms < NP; ms += chunkRows) {
        int rows = (int)((NP - ms < chunkRows) ? (NP - ms) : chunkRows);
        sampler_kernel<<<rows, 256, 0, stream>>>(f1, f2, f3, f4, av, samp, (int)ms);
        int mt = rows / 16;
        wmma_gemm_kernel<true, true><<<(mt + 7) / 8, 256, 0, stream>>>(
            samp, K1, Pbot, bb, K1, mt, nullptr, xb + (size_t)ms * K2P, K2P);
    }

    // --- 8 graph-conv layers ---
    const int mtAll = NP / 16;                       // 2561 strips
    const int gblk  = (mtAll + 7) / 8;
    const int rblk  = (int)(((size_t)NP * HID + 255) / 256);
    for (int layer = 0; layer < 8; ++layer) {
        const __bf16* Pw0; const __bf16* Pw1;
        const float* b0; const float* b1; int K;
        if (layer == 0) { Pw0 = Pg0w0; Pw1 = Pg0w1; b0 = g0b0; b1 = g0b1; K = K2P; }
        else {
            int i = layer - 1;
            Pw0 = Pgw0 + (size_t)i * HID * HID; Pw1 = Pgw1 + (size_t)i * HID * HID;
            b0 = gb0 + (size_t)i * HID;        b1 = gb1 + (size_t)i * HID; K = HID;
        }
        wmma_gemm_kernel<false, false><<<gblk, 256, 0, stream>>>(
            xb, K2P, Pw0, b0, K, mtAll, Y,  nullptr, HID);
        wmma_gemm_kernel<false, false><<<gblk, 256, 0, stream>>>(
            xb, K2P, Pw1, b1, K, mtAll, H1, nullptr, HID);
        edge_scatter_kernel<<<EDGES, HID, 0, stream>>>(edges, H1, Y);
        relu_store_kernel<<<rblk, 256, 0, stream>>>(Y, xb, xf);
    }

    // --- final 128 -> 3 projection ---
    final_proj_kernel<<<(NN + 255) / 256, 256, 0, stream>>>(xf, offw, offb, out);
}
